// TKNet_43439299231761
// MI455X (gfx1250) — compile-verified, matrix operand order checked by reading, not run on hardware
//
#include <hip/hip_runtime.h>

// ---------------------------------------------------------------------------
// CDNA5 (gfx1250) implementation of the 2s-AGCN reference network.
// All channel-mixing einsums / convs are lowered to v_wmma_f32_16x16x32_f16
// tiles (fp16 operands staged through LDS, fp32 accumulation).  Elementwise
// BN / softmax / pooling are VALU kernels.
//
// Round-3 changes: fragment-major LDS tiles (A as [m][k], B as [n][k] i.e.
// B^T) so every WMMA fragment is two contiguous 16-byte runs -> 2x
// ds_load_b128 per fragment, and staging writes are 2x ds_store_b128 per lane
// built from v8h registers.  No scalar ds_load_u16 / v_mov_b16 storm.
// ---------------------------------------------------------------------------

typedef __attribute__((ext_vector_type(16))) _Float16 v16h;
typedef __attribute__((ext_vector_type(8)))  _Float16 v8h;
typedef __attribute__((ext_vector_type(8)))  float    v8f;

#define BN_EPS 1e-5f

// ---------------------------------------------------------------------------
// Fragment load from a fragment-major LDS tile S[16][32] (row-major, 32 f16
// per row).  Per CDNA5 ISA, lane L of a 16x32 f16 A operand (and,
// by transpose symmetry, the B operand of a 32x16 tile stored as B^T) holds
// row (L&15) and K elements { kh..kh+7, 16+kh..16+kh+7 } with kh=(L>>4)*8:
// two contiguous 8-half (16B) runs -> two ds_load_b128.
// ---------------------------------------------------------------------------
__device__ __forceinline__ v16h frag_load(const _Float16* S, int row, int kh) {
  const v8h lo = *(const v8h*)(S + row * 32 + kh);
  const v8h hi = *(const v8h*)(S + row * 32 + 16 + kh);
  return __builtin_shufflevector(lo, hi, 0, 1, 2, 3, 4, 5, 6, 7,
                                 8, 9, 10, 11, 12, 13, 14, 15);
}

// ---------------------------------------------------------------------------
// Generic strided batched GEMM:  O(m,n) (+)= sum_k A(m,k) * B(k,n) + bias[m]
//   A(m,k) at A + z*aB + m*sAm + k*sAk
//   B(k,n) at B + z*bB + k*sBk + (n/colV)*colRS + (n%colV)*sBn
//   O(m,n) at O + z*oB + m*sOm + n*sOn
// One wave per 16x32 output tile (two 16x16 subtiles share the A fragment);
// K consumed in chunks of 32 through LDS.
// ---------------------------------------------------------------------------
struct GemmP {
  const float* A; const float* B; float* O; const float* bias;
  long aB, bB, oB;
  int sAm, sAk, sBk, colV, colRS, sBn, sOm, sOn;
  int M, N, K, beta;
};

__global__ __launch_bounds__(32) void k_gemm(GemmP p) {
  __shared__ alignas(16) _Float16 As[512];    // [m][k]   16x32
  __shared__ alignas(16) _Float16 Bs[1024];   // [n][k]^T two 16x32 subtiles
  const int lane = threadIdx.x;
  const int n0 = blockIdx.x * 32, m0 = blockIdx.y * 16;
  const float* Ab = p.A + (long)blockIdx.z * p.aB;
  const float* Bb = p.B + (long)blockIdx.z * p.bB;
  float* Ob = p.O + (long)blockIdx.z * p.oB;
  const int col = lane & 15, half = lane >> 4;
  const int kh = half * 8;

  // per-lane invariants ------------------------------------------------------
  // Staging: lane owns LDS row `col`, K-segment `half` (elements seg*16+j).
  const int nB0 = n0 + col;
  const int nB1 = nB0 + 16;
  const bool vn0 = nB0 < p.N;
  const bool vn1 = nB1 < p.N;
  const int nc0 = vn0 ? nB0 : (p.N - 1);
  const int nc1 = vn1 ? nB1 : (p.N - 1);
  const long cm0 = (long)(nc0 / p.colV) * p.colRS + (long)(nc0 % p.colV) * p.sBn;
  const long cm1 = (long)(nc1 / p.colV) * p.colRS + (long)(nc1 % p.colV) * p.sBn;
  const int mA = m0 + col;                 // A row staged by this lane
  const bool vmA = mA < p.M;
  const float* apA = Ab + (long)(vmA ? mA : (p.M - 1)) * p.sAm;

  v8f acc0 = {}, acc1 = {};
  if (p.beta) {
#pragma unroll
    for (int r = 0; r < 8; ++r) {
      const int m = m0 + half * 8 + r;
      if (m < p.M) {
        if (vn0) acc0[r] = Ob[(long)m * p.sOm + (long)nB0 * p.sOn];
        if (vn1) acc1[r] = Ob[(long)m * p.sOm + (long)nB1 * p.sOn];
      }
    }
  }

  for (int k0 = 0; k0 < p.K; k0 += 32) {
    {  // ---- stage A row (two 16B vector stores) ----
      v8h lo = {}, hi = {};
#pragma unroll
      for (int j = 0; j < 8; ++j) {
        const int ka = k0 + half * 16 + j;
        const int kb = ka + 8;
        const bool va = ka < p.K, vb = kb < p.K;
        const float fa = apA[(long)(va ? ka : (p.K - 1)) * p.sAk];
        const float fb = apA[(long)(vb ? kb : (p.K - 1)) * p.sAk];
        lo[j] = (_Float16)((va && vmA) ? fa : 0.f);
        hi[j] = (_Float16)((vb && vmA) ? fb : 0.f);
      }
      *(v8h*)(As + col * 32 + half * 16) = lo;
      *(v8h*)(As + col * 32 + half * 16 + 8) = hi;
    }
    {  // ---- stage B^T rows for both subtiles ----
      v8h lo0 = {}, hi0 = {}, lo1 = {}, hi1 = {};
#pragma unroll
      for (int j = 0; j < 8; ++j) {
        const int ka = k0 + half * 16 + j;
        const int kb = ka + 8;
        const bool va = ka < p.K, vb = kb < p.K;
        const long ra = (long)(va ? ka : (p.K - 1)) * p.sBk;
        const long rb = (long)(vb ? kb : (p.K - 1)) * p.sBk;
        const float a0 = Bb[ra + cm0], b0 = Bb[rb + cm0];
        const float a1 = Bb[ra + cm1], b1 = Bb[rb + cm1];
        lo0[j] = (_Float16)((va && vn0) ? a0 : 0.f);
        hi0[j] = (_Float16)((vb && vn0) ? b0 : 0.f);
        lo1[j] = (_Float16)((va && vn1) ? a1 : 0.f);
        hi1[j] = (_Float16)((vb && vn1) ? b1 : 0.f);
      }
      *(v8h*)(Bs + col * 32 + half * 16) = lo0;
      *(v8h*)(Bs + col * 32 + half * 16 + 8) = hi0;
      *(v8h*)(Bs + 512 + col * 32 + half * 16) = lo1;
      *(v8h*)(Bs + 512 + col * 32 + half * 16 + 8) = hi1;
    }
    __syncthreads();
    {
      const v16h a  = frag_load(As, col, kh);
      const v16h b0 = frag_load(Bs, col, kh);
      const v16h b1 = frag_load(Bs + 512, col, kh);
      acc0 = __builtin_amdgcn_wmma_f32_16x16x32_f16(false, a, false, b0,
                                                    (short)0, acc0, false, false);
      acc1 = __builtin_amdgcn_wmma_f32_16x16x32_f16(false, a, false, b1,
                                                    (short)0, acc1, false, false);
    }
    __syncthreads();
  }

#pragma unroll
  for (int r = 0; r < 8; ++r) {
    const int m = m0 + half * 8 + r;
    if (m < p.M) {
      const float bi = p.bias ? p.bias[m] : 0.f;
      if (vn0) Ob[(long)m * p.sOm + (long)nB0 * p.sOn] = acc0[r] + bi;
      if (vn1) Ob[(long)m * p.sOm + (long)nB1 * p.sOn] = acc1[r] + bi;
    }
  }
}

// ---------------------------------------------------------------------------
// Temporal conv (9x1, pad 4, stride s) as im2col GEMM:
//   O(o, t',v) = sum_{c,kt} W[o, c*9+kt] * X[c, t'*s + kt - 4, v]
// ---------------------------------------------------------------------------
struct TcnP {
  const float* W; const float* X; float* O;
  long xB, oB;
  int cout, cin, T, Tout, V, str;
};

__global__ __launch_bounds__(32) void k_tcn(TcnP p) {
  __shared__ alignas(16) _Float16 As[512];   // [m][k]
  __shared__ alignas(16) _Float16 Bs[512];   // [n][k]^T
  const int lane = threadIdx.x;
  const int n0 = blockIdx.x * 16, m0 = blockIdx.y * 16;
  const float* Xb = p.X + (long)blockIdx.z * p.xB;
  float* Ob = p.O + (long)blockIdx.z * p.oB;
  const int K = p.cin * 9, N = p.Tout * p.V;
  const int col = lane & 15, half = lane >> 4;
  const int kh = half * 8;

  // per-lane invariants
  const int nn = n0 + col;
  const bool vn = nn < N;
  const int nc = vn ? nn : (N - 1);
  const int vv = nc % p.V;
  const int tb = (nc / p.V) * p.str - 4;  // t_in = tb + kt
  const int mA = m0 + col;
  const bool vmA = mA < p.cout;
  const float* wpA = p.W + (long)(vmA ? mA : (p.cout - 1)) * K;

  v8f acc = {};
  for (int k0 = 0; k0 < K; k0 += 32) {
    {  // ---- stage A: W is dense row-major (cout x K) ----
      v8h lo = {}, hi = {};
#pragma unroll
      for (int j = 0; j < 8; ++j) {
        const int ka = k0 + half * 16 + j;
        const int kb = ka + 8;
        const bool va = ka < K, vb = kb < K;
        const float fa = wpA[va ? ka : (K - 1)];
        const float fb = wpA[vb ? kb : (K - 1)];
        lo[j] = (_Float16)((va && vmA) ? fa : 0.f);
        hi[j] = (_Float16)((vb && vmA) ? fb : 0.f);
      }
      *(v8h*)(As + col * 32 + half * 16) = lo;
      *(v8h*)(As + col * 32 + half * 16 + 8) = hi;
    }
    {  // ---- stage B^T: gather X[c, tb+kt, vv] ----
      v8h lo = {}, hi = {};
#pragma unroll
      for (int j = 0; j < 8; ++j) {
        const int ka = k0 + half * 16 + j;
        const int kb = ka + 8;
        {
          const bool vk = ka < K;
          const int kc = vk ? ka : (K - 1);
          const int c = kc / 9, kt = kc - 9 * c;
          const int tt = tb + kt;
          const bool vt = (tt >= 0) && (tt < p.T);
          const float f = Xb[((long)c * p.T + (vt ? tt : 0)) * p.V + vv];
          lo[j] = (_Float16)((vk && vn && vt) ? f : 0.f);
        }
        {
          const bool vk = kb < K;
          const int kc = vk ? kb : (K - 1);
          const int c = kc / 9, kt = kc - 9 * c;
          const int tt = tb + kt;
          const bool vt = (tt >= 0) && (tt < p.T);
          const float f = Xb[((long)c * p.T + (vt ? tt : 0)) * p.V + vv];
          hi[j] = (_Float16)((vk && vn && vt) ? f : 0.f);
        }
      }
      *(v8h*)(Bs + col * 32 + half * 16) = lo;
      *(v8h*)(Bs + col * 32 + half * 16 + 8) = hi;
    }
    __syncthreads();
    {
      const v16h a = frag_load(As, col, kh);
      const v16h b = frag_load(Bs, col, kh);
      acc = __builtin_amdgcn_wmma_f32_16x16x32_f16(false, a, false, b,
                                                   (short)0, acc, false, false);
    }
    __syncthreads();
  }
#pragma unroll
  for (int r = 0; r < 8; ++r) {
    const int m = m0 + half * 8 + r;
    if (m < p.cout && vn) Ob[(long)m * N + nn] = acc[r];
  }
}

// ---------------------------------------------------------------------------
// Elementwise kernels
// ---------------------------------------------------------------------------

// data_bn over (m,v,c) channels + reshape (N,C,T,V,M) -> (N*M, C, T, V)
__global__ void k_data_bn(const float* __restrict__ x, float* __restrict__ o,
                          const float* g, const float* b, const float* mu,
                          const float* var, long total) {
  long i = (long)blockIdx.x * blockDim.x + threadIdx.x;
  if (i >= total) return;
  int vv = (int)(i % 25);
  long r = i / 25;
  int t = (int)(r % 300); r /= 300;
  int c = (int)(r % 3);   r /= 3;
  int m = (int)(r & 1);
  long n = r >> 1;
  long in = (((n * 3 + c) * 300 + t) * 25 + vv) * 2 + m;
  int ch = (m * 25 + vv) * 3 + c;
  o[i] = (x[in] - mu[ch]) * (g[ch] * rsqrtf(var[ch] + BN_EPS)) + b[ch];
}

// o = [relu]( BN(x + preb[c]) + res )    (preb/res nullable; x may alias o)
__global__ void k_bn_eltwise(const float* __restrict__ x,
                             const float* __restrict__ res,
                             const float* __restrict__ preb,
                             float* __restrict__ o,
                             const float* g, const float* b, const float* mu,
                             const float* var, int C, long P, long total,
                             int relu) {
  long i = (long)blockIdx.x * blockDim.x + threadIdx.x;
  if (i >= total) return;
  int c = (int)((i / P) % C);
  float v = x[i];
  if (preb) v += preb[c];
  v = (v - mu[c]) * (g[c] * rsqrtf(var[c] + BN_EPS)) + b[c];
  if (res) v += res[i];
  if (relu) v = fmaxf(v, 0.f);
  o[i] = v;
}

// in-place: att[n,s,:,w] = softmax_v(att*scale) + A[s,:,w] + PA[s,:,w]
__global__ void k_softmax_add(float* att, const float* A, const float* PA,
                              float scale) {
  long i = (long)blockIdx.x * blockDim.x + threadIdx.x;
  if (i >= 512L * 3 * 25) return;
  int w = (int)(i % 25);
  long nms = i / 25;
  int s = (int)(nms % 3);
  float* colp = att + nms * 625 + w;
  float mx = -1e30f;
  for (int v = 0; v < 25; ++v) mx = fmaxf(mx, colp[v * 25] * scale);
  float e[25], sum = 0.f;
  for (int v = 0; v < 25; ++v) {
    e[v] = __expf(colp[v * 25] * scale - mx);
    sum += e[v];
  }
  const float inv = 1.f / sum;
  const float* Ap = A + s * 625 + w;
  const float* Pp = PA + s * 625 + w;
  for (int v = 0; v < 25; ++v) colp[v * 25] = e[v] * inv + Ap[v * 25] + Pp[v * 25];
}

// bdsum[o] = sum_s bd[s][o]
__global__ void k_sum3(const float* bd, float* o, int c) {
  int i = threadIdx.x;
  if (i < c) o[i] = bd[i] + bd[c + i] + bd[2 * c + i];
}

// pooled[n][c] = mean over m,t,v of x[(n*2+m), c, t, v]
__global__ void k_pool(const float* __restrict__ x, float* __restrict__ o,
                       int TV) {
  int i = blockIdx.x * blockDim.x + threadIdx.x;
  if (i >= 256 * 32) return;
  int c = i % 32, n = i / 32;
  float s = 0.f;
  for (int m = 0; m < 2; ++m) {
    const float* p = x + ((long)(n * 2 + m) * 32 + c) * TV;
    for (int j = 0; j < TV; ++j) s += p[j];
  }
  o[(long)n * 32 + c] = s / (2.0f * (float)TV);
}

// ---------------------------------------------------------------------------
// Host side
// ---------------------------------------------------------------------------
static void gemm(hipStream_t st, const float* A, long aB, int sAm, int sAk,
                 const float* B, long bB, int sBk, int colV, int colRS, int sBn,
                 float* O, long oB, int sOm, int sOn, const float* bias, int M,
                 int N, int K, int nb, int beta) {
  GemmP p{A, B, O, bias, aB, bB, oB, sAm, sAk, sBk,
          colV, colRS, sBn, sOm, sOn, M, N, K, beta};
  dim3 gr((unsigned)((N + 31) / 32), (unsigned)((M + 15) / 16), (unsigned)nb);
  hipLaunchKernelGGL(k_gemm, gr, dim3(32), 0, st, p);
}

struct LPar {
  const float *wa, *ba, *wb, *bb, *wd, *bd, *PA;
  const float *gg, *gb, *gm, *gv;
  const float *tw, *tb;
  const float *tg, *tbb, *tm, *tv;
  const float *dw, *db2, *dg, *dbb2, *dm, *dv;
  const float *rw, *rb, *rg, *rbb, *rm, *rv;
};

extern "C" void kernel_launch(void* const* d_in, const int* in_sizes, int n_in,
                              void* d_out, int out_size, void* d_ws,
                              size_t ws_size, hipStream_t stream) {
  (void)in_sizes; (void)n_in; (void)out_size; (void)ws_size;

  static const int  Lcin[10]  = {3, 8, 8, 8, 8, 16, 16, 16, 32, 32};
  static const int  Lcout[10] = {8, 8, 8, 8, 16, 16, 16, 32, 32, 32};
  static const int  Lstr[10]  = {1, 1, 1, 1, 2, 1, 1, 2, 1, 1};
  static const bool Lres[10]  = {false, true, true, true, true,
                                 true,  true, true, true, true};

  // ---- unpack inputs in setup_inputs() dict insertion order ----
  int idx = 0;
  auto F = [&]() -> const float* { return (const float*)d_in[idx++]; };
  const float* xin = F();                               // x
  const float* Aadj = F();                              // params.A
  const float *dg_ = F(), *db_ = F(), *dm_ = F(), *dv_ = F();  // data_bn g,b,m,v
  LPar L[10] = {};
  for (int i = 0; i < 10; ++i) {
    L[i].wa = F(); L[i].ba = F();
    L[i].wb = F(); L[i].bb = F();
    L[i].wd = F(); L[i].bd = F();
    L[i].PA = F();
    L[i].gg = F(); L[i].gb = F(); L[i].gm = F(); L[i].gv = F();
    L[i].tw = F(); L[i].tb = F();
    L[i].tg = F(); L[i].tbb = F(); L[i].tm = F(); L[i].tv = F();
    const bool down = (Lcin[i] != Lcout[i]);
    const bool resc = Lres[i] && (down || Lstr[i] != 1);
    if (down) {
      L[i].dw = F(); L[i].db2 = F();
      L[i].dg = F(); L[i].dbb2 = F(); L[i].dm = F(); L[i].dv = F();
    }
    if (resc) {
      L[i].rw = F(); L[i].rb = F();
      L[i].rg = F(); L[i].rbb = F(); L[i].rm = F(); L[i].rv = F();
    }
  }
  const float* fcw = F();
  const float* fcb = F();

  // ---- workspace carve: every layer activation is 512*8*300*25 floats ----
  const size_t BIG = (size_t)512 * 8 * 300 * 25;  // 30,720,000 floats
  float* bX   = (float*)d_ws;       // current activation (NM, C, T, V)
  float* bA   = bX + BIG;           // a-embed / u / gcn output
  float* bB   = bA + BIG;           // b-embed / y accumulator
  float* bR   = bB + BIG;           // down / block residual
  float* bT   = bR + BIG;           // tcn output
  float* bATT = bT + BIG;           // (NM, 3, 25, 25)
  float* bdsum  = bATT + (size_t)512 * 3 * 25 * 25;
  float* pooled = bdsum + 64;       // (256, 32)

  const int NB = 512;               // N*M samples

  // ---- data_bn + reshape ----
  {
    long tot = 512L * 3 * 300 * 25;
    k_data_bn<<<dim3((unsigned)((tot + 255) / 256)), dim3(256), 0, stream>>>(
        xin, bX, dg_, db_, dm_, dv_, tot);
  }

  int T = 300;
  for (int li = 0; li < 10; ++li) {
    const int cin = Lcin[li], cout = Lcout[li], str = Lstr[li];
    const int inter = cout / 4;
    const int P = T * 25;
    const long xB = (long)cin * P;

    // (1) attention embeddings a,b: per subset 1x1 conv over channels
    for (int s = 0; s < 3; ++s) {
      gemm(stream, L[li].wa + (long)s * inter * cin, 0, cin, 1,
           bX, xB, P, P, 0, 1,
           bA + (long)s * inter * P, (long)3 * inter * P, P, 1,
           L[li].ba + (long)s * inter, inter, P, cin, NB, 0);
      gemm(stream, L[li].wb + (long)s * inter * cin, 0, cin, 1,
           bX, xB, P, P, 0, 1,
           bB + (long)s * inter * P, (long)3 * inter * P, P, 1,
           L[li].bb + (long)s * inter, inter, P, cin, NB, 0);
    }
    // (2) att logits: (25 x inter*T) . (inter*T x 25)
    for (int s = 0; s < 3; ++s)
      gemm(stream, bA + (long)s * inter * P, (long)3 * inter * P, 1, 25,
           bB + (long)s * inter * P, (long)3 * inter * P, 25, 25, 0, 1,
           bATT + s * 625, 3 * 625, 25, 1,
           nullptr, 25, 25, inter * T, NB, 0);
    // (3) softmax over v + (A + PA)
    {
      long th = 512L * 3 * 25;
      k_softmax_add<<<dim3((unsigned)((th + 255) / 256)), dim3(256), 0,
                      stream>>>(bATT, Aadj, L[li].PA,
                                1.0f / (float)(inter * T));
    }
    // (4) bd.sum(0)
    k_sum3<<<1, 64, 0, stream>>>(L[li].bd, bdsum, cout);

    // (5) gcn residual (down path or identity)
    const float* gres = bX;
    const bool down = (cin != cout);
    if (down) {
      gemm(stream, L[li].dw, 0, cin, 1,
           bX, xB, P, P, 0, 1,
           bR, (long)cout * P, P, 1,
           L[li].db2, cout, P, cin, NB, 0);
      long tot = (long)NB * cout * P;
      k_bn_eltwise<<<dim3((unsigned)((tot + 255) / 256)), dim3(256), 0,
                     stream>>>(bR, nullptr, nullptr, bR, L[li].dg, L[li].dbb2,
                               L[li].dm, L[li].dv, cout, P, tot, 0);
      gres = bR;
    }

    // (6) graph conv, reassociated: u = wd_s . x ; y += u . att_s
    for (int s = 0; s < 3; ++s) {
      gemm(stream, L[li].wd + (long)s * cout * cin, 0, cin, 1,
           bX, xB, P, P, 0, 1,
           bA, (long)cout * P, P, 1,
           nullptr, cout, P, cin, NB, 0);
      gemm(stream, bA, (long)cout * P, 25, 1,
           bATT + s * 625, 3 * 625, 25, 25, 0, 1,
           bB, (long)cout * P, 25, 1,
           nullptr, cout * T, 25, 25, NB, s > 0 ? 1 : 0);
    }
    // (7) gcn out: g = relu(BN(y + bdsum) + gres)  -> bA
    {
      long tot = (long)NB * cout * P;
      k_bn_eltwise<<<dim3((unsigned)((tot + 255) / 256)), dim3(256), 0,
                     stream>>>(bB, gres, bdsum, bA, L[li].gg, L[li].gb,
                               L[li].gm, L[li].gv, cout, P, tot, 1);
    }

    const int Tout = (T - 1) / str + 1;
    const int Pout = Tout * 25;

    // (8) block residual
    const float* r2 = nullptr;
    const bool resc = Lres[li] && (down || str != 1);
    if (resc) {
      gemm(stream, L[li].rw, 0, cin, 1,
           bX, xB, P, 25, str * 25, 1,       // strided column map over T
           bR, (long)cout * Pout, Pout, 1,
           L[li].rb, cout, Pout, cin, NB, 0);
      long tot = (long)NB * cout * Pout;
      k_bn_eltwise<<<dim3((unsigned)((tot + 255) / 256)), dim3(256), 0,
                     stream>>>(bR, nullptr, nullptr, bR, L[li].rg, L[li].rbb,
                               L[li].rm, L[li].rv, cout, Pout, tot, 0);
      r2 = bR;
    } else if (Lres[li]) {
      r2 = bX;  // identity residual (stride 1, cin==cout)
    }

    // (9) temporal conv (im2col WMMA GEMM)
    {
      TcnP tp{L[li].tw, bA, bT, (long)cout * P, (long)cout * Pout,
              cout, cout, T, Tout, 25, str};
      dim3 gr((unsigned)((Pout + 15) / 16), (unsigned)((cout + 15) / 16),
              (unsigned)NB);
      hipLaunchKernelGGL(k_tcn, gr, dim3(32), 0, stream, tp);
    }
    // (10) block out: x = relu(BN(t + tcn_b) + r2) -> bX
    {
      long tot = (long)NB * cout * Pout;
      k_bn_eltwise<<<dim3((unsigned)((tot + 255) / 256)), dim3(256), 0,
                     stream>>>(bT, r2, L[li].tb, bX, L[li].tg, L[li].tbb,
                               L[li].tm, L[li].tv, cout, Pout, tot, 1);
    }
    T = Tout;
  }

  // ---- global average pool over (m, t, v) ----
  k_pool<<<dim3((256 * 32 + 255) / 256), dim3(256), 0, stream>>>(bX, pooled,
                                                                 T * 25);
  // ---- FC: out[n, cls] = fc_w[cls,:] . pooled[n,:] + fc_b[cls] ----
  gemm(stream, fcw, 0, 32, 1,
       pooled, 0, 1, 256, 0, 32,
       (float*)d_out, 0, 1, 60,
       fcb, 60, 256, 32, 1, 0);
}
